// Transformer_Block_41025527611409
// MI455X (gfx1250) — compile-verified, matrix operand order checked by reading, not run on hardware
//
#include <hip/hip_runtime.h>

typedef __attribute__((ext_vector_type(16))) _Float16 v16h;
typedef __attribute__((ext_vector_type(8)))  _Float16 half8;
typedef __attribute__((ext_vector_type(8)))  float    v8f;

#define BATCH 32
#define SEQ   1024
#define CH    256
#define NHEAD 4
#define HDIM  64
#define FFN   1024
#define MTOK  (BATCH*SEQ)     // 32768 tokens
#define EPSV  1e-5f

// ---- fragment load: two contiguous 16B chunks at +0 and +16 halves ----
static __device__ __forceinline__ v16h load_frag(const _Float16* p) {
  half8 lo = *(const half8*)p;
  half8 hi = *(const half8*)(p + 16);
  return __builtin_shufflevector(lo, hi, 0,1,2,3,4,5,6,7,8,9,10,11,12,13,14,15);
}

// ---------------- f32 -> f16 convert ----------------
__global__ void cvt16_kernel(const float* __restrict__ in, _Float16* __restrict__ out, int n) {
  int i = blockIdx.x * blockDim.x + threadIdx.x;
  if (i < n) out[i] = (_Float16)in[i];
}

// ---------------- BatchNorm stats: one block per channel ----------------
__global__ void bn_stats_kernel(const float* __restrict__ x, float* __restrict__ mean,
                                float* __restrict__ rstd) {
  int c = blockIdx.x;
  int t = threadIdx.x;
  float s = 0.f, s2 = 0.f;
  for (int i = t; i < MTOK; i += 256) {
    float v = x[(size_t)i * CH + c];
    s += v; s2 += v * v;
  }
  __shared__ float sh[256], sh2[256];
  sh[t] = s; sh2[t] = s2;
  __syncthreads();
  for (int off = 128; off > 0; off >>= 1) {
    if (t < off) { sh[t] += sh[t + off]; sh2[t] += sh2[t + off]; }
    __syncthreads();
  }
  if (t == 0) {
    float m = sh[0] / (float)MTOK;
    float var = sh2[0] / (float)MTOK - m * m;
    mean[c] = m;
    rstd[c] = rsqrtf(var + EPSV);
  }
}

// ---------------- BN apply -> f16 ----------------
__global__ void bn_apply_kernel(const float* __restrict__ x, const float* __restrict__ mean,
                                const float* __restrict__ rstd, const float* __restrict__ g,
                                const float* __restrict__ beta, _Float16* __restrict__ out) {
  int i = blockIdx.x * 256 + threadIdx.x;
  int c = i & (CH - 1);
  out[i] = (_Float16)((x[i] - mean[c]) * rstd[c] * g[c] + beta[c]);
}

// ---------------- transpose V: (b,n,h,d) f16 -> (b,h,d,n) f16 ----------------
__global__ void vtrans_kernel(const _Float16* __restrict__ v, _Float16* __restrict__ vt) {
  int i = blockIdx.x * 256 + threadIdx.x;     // over MTOK*CH
  int key = i & (SEQ - 1);
  int d   = (i >> 10) & (HDIM - 1);
  int bh  = i >> 16;
  int b = bh >> 2, h = bh & 3;
  vt[i] = v[(size_t)((b << 10) + key) * CH + h * HDIM + d];
}

// ---------------- generic WMMA GEMM: out[M,N] = A[M,K] @ Bw[N,K]^T + bias ----------------
// wave computes a 32x64 tile (two 16-row subtiles share every B fragment);
// optional residual (stride N), relu, f16/f32 outputs.
__global__ void gemm_wmma_kernel(const _Float16* __restrict__ A, const _Float16* __restrict__ Bw,
                                 const float* __restrict__ bias, const float* __restrict__ resid,
                                 _Float16* __restrict__ out16, float* __restrict__ out32,
                                 int K, int N, int Mtiles32, int relu_flag) {
  int wave = blockIdx.x * (blockDim.x >> 5) + (threadIdx.x >> 5);
  int lane = threadIdx.x & 31;
  int mt = wave % Mtiles32;
  int nt = wave / Mtiles32;
  int m0 = mt * 32, n0 = nt * 64;
  int l16 = lane & 15, hi = lane >> 4;
  int koff = hi * 8;

  const _Float16* arow0 = A + (size_t)(m0 + l16) * K + koff;
  const _Float16* arow1 = arow0 + (size_t)16 * K;
  v8f z = {};
  v8f acc0[4] = { z, z, z, z };   // rows m0..m0+15
  v8f acc1[4] = { z, z, z, z };   // rows m0+16..m0+31

  for (int k0 = 0; k0 < K; k0 += 32) {
    v16h a0 = load_frag(arow0 + k0);
    v16h a1 = load_frag(arow1 + k0);
#pragma unroll
    for (int t = 0; t < 4; ++t) {
      const _Float16* brow = Bw + (size_t)(n0 + t * 16 + l16) * K + k0 + koff;
      v16h b = load_frag(brow);
      acc0[t] = __builtin_amdgcn_wmma_f32_16x16x32_f16(false, a0, false, b, (short)0, acc0[t],
                                                       false, false);
      acc1[t] = __builtin_amdgcn_wmma_f32_16x16x32_f16(false, a1, false, b, (short)0, acc1[t],
                                                       false, false);
    }
  }

#pragma unroll
  for (int t = 0; t < 4; ++t) {
    int n = n0 + t * 16 + l16;
    float bn = bias ? bias[n] : 0.f;
#pragma unroll
    for (int r = 0; r < 8; ++r) {
      int m = m0 + hi * 8 + r;
      float v0 = acc0[t][r] + bn;
      float v1 = acc1[t][r] + bn;
      if (resid) {
        v0 += resid[(size_t)m * N + n];
        v1 += resid[(size_t)(m + 16) * N + n];
      }
      if (relu_flag) { v0 = fmaxf(v0, 0.f); v1 = fmaxf(v1, 0.f); }
      if (out16) {
        out16[(size_t)m * N + n]        = (_Float16)v0;
        out16[(size_t)(m + 16) * N + n] = (_Float16)v1;
      }
      if (out32) {
        out32[(size_t)m * N + n]        = v0;
        out32[(size_t)(m + 16) * N + n] = v1;
      }
    }
  }
}

// ---------------- flash attention: wave = 16 queries of one (b,h) ----------------
__global__ void attn_kernel(const _Float16* __restrict__ q, const _Float16* __restrict__ k,
                            const _Float16* __restrict__ vt, _Float16* __restrict__ o) {
  __shared__ __align__(16) _Float16 pbuf[4][16 * 32];
  int wave = blockIdx.x * 4 + (threadIdx.x >> 5);
  int lane = threadIdx.x & 31;
  int wib  = threadIdx.x >> 5;
  int bh = wave >> 6;          // 64 query tiles per (b,h)
  int qt = wave & 63;
  int b = bh >> 2, h = bh & 3;
  int l16 = lane & 15, hi = lane >> 4, koff = hi * 8;

  const _Float16* qrow = q + (size_t)((b << 10) + qt * 16 + l16) * CH + h * HDIM;
  v16h qa0 = load_frag(qrow + koff);        // d = 0..31 slice
  v16h qa1 = load_frag(qrow + 32 + koff);   // d = 32..63 slice

  v8f z = {};
  v8f acc[4] = { z, z, z, z };
  float mrow[8], lrow[8];
#pragma unroll
  for (int r = 0; r < 8; ++r) { mrow[r] = -1e30f; lrow[r] = 0.f; }

  _Float16* pw = pbuf[wib];

  for (int kb = 0; kb < SEQ; kb += 32) {
    v8f s0 = z, s1 = z;
    {
      const _Float16* krow0 = k + (size_t)((b << 10) + kb + l16) * CH + h * HDIM;
      v16h kf0a = load_frag(krow0 + koff);
      v16h kf0b = load_frag(krow0 + 32 + koff);
      s0 = __builtin_amdgcn_wmma_f32_16x16x32_f16(false, qa0, false, kf0a, (short)0, s0, false, false);
      s0 = __builtin_amdgcn_wmma_f32_16x16x32_f16(false, qa1, false, kf0b, (short)0, s0, false, false);
      const _Float16* krow1 = k + (size_t)((b << 10) + kb + 16 + l16) * CH + h * HDIM;
      v16h kf1a = load_frag(krow1 + koff);
      v16h kf1b = load_frag(krow1 + 32 + koff);
      s1 = __builtin_amdgcn_wmma_f32_16x16x32_f16(false, qa0, false, kf1a, (short)0, s1, false, false);
      s1 = __builtin_amdgcn_wmma_f32_16x16x32_f16(false, qa1, false, kf1b, (short)0, s1, false, false);
    }
    // online softmax over the 32 new keys; row = r + 8*hi, cols spread over 16 lanes of half
#pragma unroll
    for (int r = 0; r < 8; ++r) {
      float a0 = s0[r] * 0.125f;   // 1/sqrt(64)
      float a1 = s1[r] * 0.125f;
      float mx = fmaxf(a0, a1);
      for (int off = 1; off < 16; off <<= 1) mx = fmaxf(mx, __shfl_xor(mx, off, 16));
      float mnew = fmaxf(mrow[r], mx);
      float alpha = __expf(mrow[r] - mnew);
      float p0 = __expf(a0 - mnew);
      float p1 = __expf(a1 - mnew);
      float ps = p0 + p1;
      for (int off = 1; off < 16; off <<= 1) ps += __shfl_xor(ps, off, 16);
      lrow[r] = lrow[r] * alpha + ps;
      mrow[r] = mnew;
#pragma unroll
      for (int t = 0; t < 4; ++t) acc[t][r] *= alpha;
      int qr = r + hi * 8;
      pw[qr * 32 + l16]      = (_Float16)p0;
      pw[qr * 32 + 16 + l16] = (_Float16)p1;
    }
    // P (16x32) as A-fragment from wave-private LDS; same-wave DS ops are in-order
    v16h pa = load_frag(pw + l16 * 32 + koff);
#pragma unroll
    for (int t = 0; t < 4; ++t) {
      const _Float16* vrow = vt + (size_t)(bh * HDIM + t * 16 + l16) * SEQ + kb + koff;
      v16h vb = load_frag(vrow);
      acc[t] = __builtin_amdgcn_wmma_f32_16x16x32_f16(false, pa, false, vb, (short)0, acc[t],
                                                      false, false);
    }
  }

#pragma unroll
  for (int t = 0; t < 4; ++t) {
#pragma unroll
    for (int r = 0; r < 8; ++r) {
      int qr = r + hi * 8;
      float val = acc[t][r] / lrow[r];
      o[(size_t)((b << 10) + qt * 16 + qr) * CH + h * HDIM + t * 16 + l16] = (_Float16)val;
    }
  }
}

// ============================================================================
extern "C" void kernel_launch(void* const* d_in, const int* in_sizes, int n_in,
                              void* d_out, int out_size, void* d_ws, size_t ws_size,
                              hipStream_t stream) {
  (void)in_sizes; (void)n_in; (void)out_size; (void)ws_size;

  const float* x    = (const float*)d_in[0];
  const float* wq   = (const float*)d_in[1];  const float* bq  = (const float*)d_in[2];
  const float* wk   = (const float*)d_in[3];  const float* bk  = (const float*)d_in[4];
  const float* wv   = (const float*)d_in[5];  const float* bv  = (const float*)d_in[6];
  const float* wo   = (const float*)d_in[7];  const float* bo  = (const float*)d_in[8];
  const float* g1   = (const float*)d_in[9];  const float* be1 = (const float*)d_in[10];
  const float* g2   = (const float*)d_in[11]; const float* be2 = (const float*)d_in[12];
  const float* w1   = (const float*)d_in[13]; const float* bf1 = (const float*)d_in[14];
  const float* w2   = (const float*)d_in[15]; const float* bf2 = (const float*)d_in[16];
  float* outp = (float*)d_out;

  // ---- workspace layout ----
  char* ws = (char*)d_ws;
  size_t off = 0;
  auto alloc = [&](size_t bytes) { char* p = ws + off; off += (bytes + 255) & ~(size_t)255; return p; };

  float*    mean1 = (float*)alloc(CH * 4);
  float*    rstd1 = (float*)alloc(CH * 4);
  float*    mean2 = (float*)alloc(CH * 4);
  float*    rstd2 = (float*)alloc(CH * 4);
  _Float16* wq16  = (_Float16*)alloc((size_t)CH * CH * 2);
  _Float16* wk16  = (_Float16*)alloc((size_t)CH * CH * 2);
  _Float16* wv16  = (_Float16*)alloc((size_t)CH * CH * 2);
  _Float16* wo16  = (_Float16*)alloc((size_t)CH * CH * 2);
  _Float16* w116  = (_Float16*)alloc((size_t)FFN * CH * 2);
  _Float16* w216  = (_Float16*)alloc((size_t)CH * FFN * 2);
  _Float16* h16   = (_Float16*)alloc((size_t)MTOK * CH * 2);   // BN1 out, later BN2 out
  _Float16* q16   = (_Float16*)alloc((size_t)MTOK * CH * 2);
  _Float16* k16   = (_Float16*)alloc((size_t)MTOK * CH * 2);
  _Float16* v16   = (_Float16*)alloc((size_t)MTOK * CH * 2);   // V, later attention output
  _Float16* vt16  = (_Float16*)alloc((size_t)MTOK * CH * 2);   // V transposed (b,h,d,n)
  float*    r1    = (float*)alloc((size_t)MTOK * CH * 4);
  _Float16* mlp16 = (_Float16*)alloc((size_t)MTOK * FFN * 2);

  // ---- weights -> f16 ----
  cvt16_kernel<<<(CH * CH + 255) / 256, 256, 0, stream>>>(wq, wq16, CH * CH);
  cvt16_kernel<<<(CH * CH + 255) / 256, 256, 0, stream>>>(wk, wk16, CH * CH);
  cvt16_kernel<<<(CH * CH + 255) / 256, 256, 0, stream>>>(wv, wv16, CH * CH);
  cvt16_kernel<<<(CH * CH + 255) / 256, 256, 0, stream>>>(wo, wo16, CH * CH);
  cvt16_kernel<<<(FFN * CH + 255) / 256, 256, 0, stream>>>(w1, w116, FFN * CH);
  cvt16_kernel<<<(CH * FFN + 255) / 256, 256, 0, stream>>>(w2, w216, CH * FFN);

  // ---- BN1 ----
  bn_stats_kernel<<<CH, 256, 0, stream>>>(x, mean1, rstd1);
  bn_apply_kernel<<<(MTOK * CH) / 256, 256, 0, stream>>>(x, mean1, rstd1, g1, be1, h16);

  // ---- QKV projections (M=32768, K=256, N=256); wave tile = 32x64 ----
  const int MT32 = MTOK / 32;                      // 1024 M-tiles
  int qkv_blocks  = (MT32 * (CH  / 64)) / 4;       // 4096 waves / 4 per block
  int mlp1_blocks = (MT32 * (FFN / 64)) / 4;       // 16384 waves / 4 per block
  gemm_wmma_kernel<<<qkv_blocks, 128, 0, stream>>>(h16, wq16, bq, nullptr, q16, nullptr,
                                                   CH, CH, MT32, 0);
  gemm_wmma_kernel<<<qkv_blocks, 128, 0, stream>>>(h16, wk16, bk, nullptr, k16, nullptr,
                                                   CH, CH, MT32, 0);
  gemm_wmma_kernel<<<qkv_blocks, 128, 0, stream>>>(h16, wv16, bv, nullptr, v16, nullptr,
                                                   CH, CH, MT32, 0);

  // ---- V transpose, flash attention (output reuses V buffer) ----
  vtrans_kernel<<<(MTOK * CH) / 256, 256, 0, stream>>>(v16, vt16);
  attn_kernel<<<(BATCH * NHEAD * (SEQ / 16)) / 4, 128, 0, stream>>>(q16, k16, vt16, v16);

  // ---- O projection + residual(x) -> r1 (f32) ----
  gemm_wmma_kernel<<<qkv_blocks, 128, 0, stream>>>(v16, wo16, bo, x, nullptr, r1,
                                                   CH, CH, MT32, 0);

  // ---- BN2 ----
  bn_stats_kernel<<<CH, 256, 0, stream>>>(r1, mean2, rstd2);
  bn_apply_kernel<<<(MTOK * CH) / 256, 256, 0, stream>>>(r1, mean2, rstd2, g2, be2, h16);

  // ---- MLP ----
  gemm_wmma_kernel<<<mlp1_blocks, 128, 0, stream>>>(h16, w116, bf1, nullptr, mlp16, nullptr,
                                                    CH, FFN, MT32, 1);
  gemm_wmma_kernel<<<qkv_blocks, 128, 0, stream>>>(mlp16, w216, bf2, r1, nullptr, outp,
                                                   FFN, CH, MT32, 0);
}
